// Encoder_74887049773725
// MI455X (gfx1250) — compile-verified, matrix operand order checked by reading, not run on hardware
//
#include <hip/hip_runtime.h>

// ---------------------------------------------------------------------------
// MI455X (gfx1250) transformer encoder, bf16 WMMA (V_WMMA_F32_16X16X32_BF16),
// fp32 accumulate.  F=100 padded to 128; all pad lanes are exact zeros so the
// padding is mathematically inert through every GEMM in the chain.
// Round 3: + deep global_prefetch_b8 (one pipeline depth ahead) on the A/B
// streams; double-buffered fragment pipeline and LDS-repacked coalesced bf16
// epilogues retained from round 2 (verified: partial loadcnt waits, ds>0).
// ---------------------------------------------------------------------------

typedef __bf16 bf16;
typedef __bf16 v8bf  __attribute__((ext_vector_type(8)));
typedef __bf16 v16bf __attribute__((ext_vector_type(16)));
typedef float  v8f   __attribute__((ext_vector_type(8)));

#define C_      32
#define T_      1024
#define F_      100
#define HEAD_   5
#define FFN_    64
#define HLP_    200
#define NPRED_  24
#define NLAYERS_ 3
#define FP_     128           // padded feature dim (K mult of 32, N mult of 16)
#define HFP_    (HEAD_ * FP_) // 640, padded concat-head dim

enum {
  EP_SCORES = 0,         // f32 out, * scale
  EP_BF16 = 1,           // bf16 out, row-major (LDS-repacked b128 stores)
  EP_BF16_T = 2,         // bf16 out, transposed store (for V^T)
  EP_BIAS_RELU_BF16 = 3, // relu(acc + bias) -> bf16
  EP_RESID_F32 = 4,      // acc + resid -> f32 (n < nvalid)
  EP_BIAS_RESID_F32 = 5  // acc + bias + resid -> f32 (n < nvalid)
};

// Load one 16x32 bf16 fragment half-pair: per CDNA5 ISA 16-bit A layout the
// lane needs K = [koff..koff+7] and [koff+16..koff+23] -> two 16B loads.
__device__ __forceinline__ v16bf ld_frag(const bf16* p) {
  const v8bf* q = (const v8bf*)p;
  const v8bf lo = q[0];
  const v8bf hi = q[2];
  v16bf r;
#pragma unroll
  for (int i = 0; i < 8; ++i) { r[i] = lo[i]; r[8 + i] = hi[i]; }
  return r;
}

// ---------------------------------------------------------------------------
// Generic bf16 WMMA GEMM:  Out[M,N](f32 acc) = A[M,K] * B[K,N]
//  * A row-major (lda = K-stride).  B stored COLUMN-contiguous: column n of B
//    lives at Bb + n*ldb + k  (weights / V are pre-transposed at cast time).
//  * Block = 256 threads = 8 waves; wave w owns a 16(M) x 64(N) strip ->
//    4 accumulators, A fragment reused 4x per K-step.
//  * K pipelined in steps of 64 with two register fragment buffers so loads
//    for step i+1 are in flight during the WMMAs of step i; the step after
//    that is pulled toward the WGP cache with global_prefetch_b8.
//  * M mult of 128, N mult of 64, K mult of 64 (all call sites satisfy).
// ---------------------------------------------------------------------------
template <int MODE>
__global__ __launch_bounds__(256) void wmma_gemm(
    const bf16* __restrict__ A, long sA, int zdivA, int lda,
    const bf16* __restrict__ B, long sB, int zmodB, int ldb,
    void* __restrict__ O, long sO, int ldo,
    const float* __restrict__ bias, const float* __restrict__ resid,
    int K, int nvalid, float scale) {
  const int wave = threadIdx.x >> 5;
  const int lane = threadIdx.x & 31;
  const int hi   = lane >> 4;   // 0 or 1 (lane half)
  const int r16  = lane & 15;
  const int m0   = blockIdx.x * 128 + wave * 16;
  const int n0   = blockIdx.y * 64;
  const int z    = blockIdx.z;

  const bf16* Ab = A + (long)(z / zdivA) * sA;
  const bf16* Bb = B + (long)(z % zmodB) * sB;

  v8f acc[4];
#pragma unroll
  for (int j = 0; j < 4; ++j)
#pragma unroll
    for (int i = 0; i < 8; ++i) acc[j][i] = 0.0f;

  const int koff = hi << 3;  // per-lane-half K offset (0 / 8)
  const bf16* pArow = Ab + (long)(m0 + r16) * lda + koff;
  const bf16* pB[4];
#pragma unroll
  for (int j = 0; j < 4; ++j) pB[j] = Bb + (long)(n0 + j * 16 + r16) * ldb + koff;

  // Prologue: fragment buffer 0 <- K-step 0.
  v16bf af0 = ld_frag(pArow);
  v16bf bf0[4];
#pragma unroll
  for (int j = 0; j < 4; ++j) bf0[j] = ld_frag(pB[j]);

  for (int kk = 0; kk < K; kk += 64) {
    // Deep prefetch: pull the K-block one pipeline depth ahead toward the
    // WGP cache (emits global_prefetch_b8; speculative, no counter traffic).
    if (kk + 128 < K) {
      __builtin_prefetch(pArow + kk + 128, 0, 1);
#pragma unroll
      for (int j = 0; j < 4; ++j) __builtin_prefetch(pB[j] + kk + 128, 0, 1);
    }

    // Issue loads for K-step kk+32 (buffer 1), then MAC buffer 0: the WMMAs
    // only need a partial s_wait_loadcnt, overlapping load latency.
    v16bf af1 = ld_frag(pArow + kk + 32);
    v16bf bf1[4];
#pragma unroll
    for (int j = 0; j < 4; ++j) bf1[j] = ld_frag(pB[j] + kk + 32);
#pragma unroll
    for (int j = 0; j < 4; ++j)
      acc[j] = __builtin_amdgcn_wmma_f32_16x16x32_bf16(
          false, af0, false, bf0[j], (short)0, acc[j], false, false);

    // Issue loads for K-step kk+64 (buffer 0), then MAC buffer 1.
    if (kk + 64 < K) {
      af0 = ld_frag(pArow + kk + 64);
#pragma unroll
      for (int j = 0; j < 4; ++j) bf0[j] = ld_frag(pB[j] + kk + 64);
    }
#pragma unroll
    for (int j = 0; j < 4; ++j)
      acc[j] = __builtin_amdgcn_wmma_f32_16x16x32_bf16(
          false, af1, false, bf1[j], (short)0, acc[j], false, false);
  }

  // ---- Epilogue.  C/D layout: VGPR v -> row m0 + 8*hi + v, col (lane&15). --
  if constexpr (MODE == EP_BF16 || MODE == EP_BIAS_RELU_BF16) {
    // Repack the wave's 16x64 tile through LDS (same-wave write->read is
    // in-order, no barrier) and emit 4 coalesced b128 stores per lane.
    __shared__ __align__(16) bf16 stile[8][16][72];  // 72: +8 pad, 16B-mult row
    bf16 (*tw)[72] = stile[wave];
#pragma unroll
    for (int j = 0; j < 4; ++j)
#pragma unroll
      for (int v = 0; v < 8; ++v) {
        float val = acc[j][v];
        if (MODE == EP_BIAS_RELU_BF16)
          val = fmaxf(val + bias[n0 + j * 16 + r16], 0.0f);
        tw[(hi << 3) + v][j * 16 + r16] = (bf16)val;
      }
    bf16* o = (bf16*)O + (long)z * sO;
#pragma unroll
    for (int t = 0; t < 4; ++t) {
      const int chunk = t * 32 + lane;  // 128 chunks of 8 bf16
      const int row = chunk >> 3;       // 0..15
      const int c8  = chunk & 7;        // 0..7
      const v8bf d = *(const v8bf*)&tw[row][c8 * 8];
      *(v8bf*)&o[(long)(m0 + row) * ldo + n0 + c8 * 8] = d;
    }
  } else if constexpr (MODE == EP_BF16_T) {
    // Transposed store (V^T): tile is 64(n) x 16(m) in LDS.
    __shared__ __align__(16) bf16 ttile[8][64][24];  // 24: 48B row, 16B-mult
    bf16 (*tw)[24] = ttile[wave];
#pragma unroll
    for (int j = 0; j < 4; ++j)
#pragma unroll
      for (int v = 0; v < 8; ++v)
        tw[j * 16 + r16][(hi << 3) + v] = (bf16)acc[j][v];
    bf16* o = (bf16*)O + (long)z * sO;
#pragma unroll
    for (int t = 0; t < 4; ++t) {
      const int chunk = t * 32 + lane;  // 128 chunks of 8 bf16
      const int row  = chunk >> 1;      // 0..63 (n)
      const int half = chunk & 1;       // 0..1  (m half)
      const v8bf d = *(const v8bf*)&tw[row][half * 8];
      *(v8bf*)&o[(long)(n0 + row) * ldo + m0 + half * 8] = d;
    }
  } else {
    // f32 epilogues: 16 consecutive lanes hit contiguous addresses already.
#pragma unroll
    for (int j = 0; j < 4; ++j) {
      const int n = n0 + j * 16 + r16;
#pragma unroll
      for (int v = 0; v < 8; ++v) {
        const int m = m0 + (hi << 3) + v;
        const float val = acc[j][v];
        if constexpr (MODE == EP_SCORES) {
          float* o = (float*)O + (long)z * sO;
          o[(long)m * ldo + n] = val * scale;
        } else if constexpr (MODE == EP_RESID_F32) {
          if (n < nvalid) {
            float* o = (float*)O;
            o[(long)m * ldo + n] = val + resid[(long)m * ldo + n];
          }
        } else if constexpr (MODE == EP_BIAS_RESID_F32) {
          if (n < nvalid) {
            float* o = (float*)O;
            o[(long)m * ldo + n] = val + bias[n] + resid[(long)m * ldo + n];
          }
        }
      }
    }
  }
}

// One wave per 1024-wide score row: max, exp, sum, normalize -> bf16 probs.
__global__ __launch_bounds__(256) void softmax_rows(
    const float* __restrict__ S, bf16* __restrict__ P) {
  const int row  = blockIdx.x * 8 + (threadIdx.x >> 5);
  const int lane = threadIdx.x & 31;
  const float* s = S + (long)row * T_;
  float v[32];
  float mx = -3.4e38f;
#pragma unroll
  for (int i = 0; i < 32; ++i) { v[i] = s[lane + 32 * i]; mx = fmaxf(mx, v[i]); }
#pragma unroll
  for (int off = 16; off > 0; off >>= 1) mx = fmaxf(mx, __shfl_xor(mx, off, 32));
  float sum = 0.0f;
#pragma unroll
  for (int i = 0; i < 32; ++i) { v[i] = __expf(v[i] - mx); sum += v[i]; }
#pragma unroll
  for (int off = 16; off > 0; off >>= 1) sum += __shfl_xor(sum, off, 32);
  const float inv = 1.0f / sum;
  bf16* p = P + (long)row * T_;
#pragma unroll
  for (int i = 0; i < 32; ++i) p[lane + 32 * i] = (bf16)(v[i] * inv);
}

// One wave per 100-feature row.
__global__ __launch_bounds__(256) void layernorm_rows(
    const float* __restrict__ in, float* __restrict__ out,
    const float* __restrict__ g, const float* __restrict__ b) {
  const int row  = blockIdx.x * 8 + (threadIdx.x >> 5);
  const int lane = threadIdx.x & 31;
  const float* x = in + (long)row * F_;
  float v[4];
  float sum = 0.0f;
#pragma unroll
  for (int i = 0; i < 4; ++i) {
    const int idx = lane + 32 * i;
    v[i] = (idx < F_) ? x[idx] : 0.0f;
    sum += v[i];
  }
#pragma unroll
  for (int off = 16; off > 0; off >>= 1) sum += __shfl_xor(sum, off, 32);
  const float mean = sum * (1.0f / F_);
  float var = 0.0f;
#pragma unroll
  for (int i = 0; i < 4; ++i) {
    const int idx = lane + 32 * i;
    const float d = (idx < F_) ? (v[i] - mean) : 0.0f;
    var += d * d;
  }
#pragma unroll
  for (int off = 16; off > 0; off >>= 1) var += __shfl_xor(var, off, 32);
  const float inv = rsqrtf(var * (1.0f / F_) + 1e-5f);
  float* o = out + (long)row * F_;
#pragma unroll
  for (int i = 0; i < 4; ++i) {
    const int idx = lane + 32 * i;
    if (idx < F_) o[idx] = (v[i] - mean) * inv * g[idx] + b[idx];
  }
}

// fp32 [rows, cin] -> bf16 [rows, cout], zero-padded columns.
__global__ void cast_pad(const float* __restrict__ src, bf16* __restrict__ dst,
                         int rows, int cin, int cout) {
  const long total = (long)rows * cout;
  const long i = (long)blockIdx.x * blockDim.x + threadIdx.x;
  if (i >= total) return;
  const int c = (int)(i % cout);
  const long r = i / cout;
  dst[i] = (c < cin) ? (bf16)src[r * cin + c] : (bf16)0.0f;
}

// fp32 [kin, nin] -> bf16 [np, kp] with dst[n][k] = src[k][n] (zero pad).
// Batched via grid.z (per-head weights).
__global__ void cast_transpose_pad(const float* __restrict__ src,
                                   bf16* __restrict__ dst, int kin, int nin,
                                   int kp, int np) {
  const float* s = src + (long)blockIdx.z * kin * nin;
  bf16* d = dst + (long)blockIdx.z * np * kp;
  const int i = blockIdx.x * blockDim.x + threadIdx.x;
  if (i >= np * kp) return;
  const int k = i % kp;
  const int n = i / kp;
  d[i] = (k < kin && n < nin) ? (bf16)s[(long)k * nin + n] : (bf16)0.0f;
}

// Wm [H*F=500, F=100] -> Wm^T padded bf16 [128][640]; the 640-dim K axis is
// per-head padded (h*128 + f, zeros for f in [100,128)) to match o-concat.
__global__ void cast_wm(const float* __restrict__ src, bf16* __restrict__ dst) {
  const int i = blockIdx.x * blockDim.x + threadIdx.x;
  if (i >= FP_ * HFP_) return;
  const int n = i / HFP_;
  const int j = i % HFP_;
  const int h = j >> 7;
  const int f = j & 127;
  dst[i] = (n < F_ && f < F_) ? (bf16)src[((long)(h * F_ + f)) * F_ + n]
                              : (bf16)0.0f;
}

// 24-step autoregressive head; tiny and strictly sequential -> one block, fp32.
__global__ __launch_bounds__(256) void predictor(
    const float* __restrict__ x, const float* __restrict__ w1,
    const float* __restrict__ b1, const float* __restrict__ w2,
    const float* __restrict__ b2, float* __restrict__ out) {
  __shared__ float cur[C_][F_];
  __shared__ float hid[C_][HLP_];
  const int tid = threadIdx.x;
  for (int i = tid; i < C_ * F_; i += 256) {
    const int c = i / F_, f = i % F_;
    cur[c][f] = x[((long)c * T_ + (T_ - 1)) * F_ + f];
  }
  __syncthreads();
  for (int p = 0; p < NPRED_; ++p) {
    for (int i = tid; i < C_ * HLP_; i += 256) {
      const int c = i / HLP_, j = i % HLP_;
      float s = b1[j];
      for (int f = 0; f < F_; ++f) s += cur[c][f] * w1[f * HLP_ + j];
      hid[c][j] = s;
    }
    __syncthreads();
    for (int i = tid; i < C_ * F_; i += 256) {
      const int c = i / F_, f = i % F_;
      float s = b2[f];
      for (int j = 0; j < HLP_; ++j) s += hid[c][j] * w2[j * F_ + f];
      out[((long)c * NPRED_ + p) * F_ + f] = s;
      cur[c][f] = s;  // cur not read in this phase
    }
    __syncthreads();
  }
}

extern "C" void kernel_launch(void* const* d_in, const int* in_sizes, int n_in,
                              void* d_out, int out_size, void* d_ws,
                              size_t ws_size, hipStream_t stream) {
  (void)in_sizes; (void)n_in; (void)out_size; (void)ws_size;
  const float* X    = (const float*)d_in[0];
  const float* WQ   = (const float*)d_in[1];
  const float* WK   = (const float*)d_in[2];
  const float* WV   = (const float*)d_in[3];
  const float* Wm   = (const float*)d_in[4];
  const float* W1   = (const float*)d_in[5];
  const float* B1   = (const float*)d_in[6];
  const float* W2   = (const float*)d_in[7];
  const float* B2   = (const float*)d_in[8];
  const float* LN1g = (const float*)d_in[9];
  const float* LN1b = (const float*)d_in[10];
  const float* LN2g = (const float*)d_in[11];
  const float* LN2b = (const float*)d_in[12];
  const float* LPw1 = (const float*)d_in[13];
  const float* LPb1 = (const float*)d_in[14];
  const float* LPw2 = (const float*)d_in[15];
  const float* LPb2 = (const float*)d_in[16];

  const long CT = (long)C_ * T_;
  char* base = (char*)d_ws;
  size_t off = 0;
  auto take = [&](size_t bytes) -> void* {
    off = (off + 255) & ~(size_t)255;
    void* p = base + off;
    off += bytes;
    return p;
  };
  float* x     = (float*)take(CT * F_ * 4);                      // activations
  bf16*  xb    = (bf16*) take(CT * FP_ * 2);                     // bf16 padded x
  bf16*  qbuf  = (bf16*) take((long)C_ * HEAD_ * T_ * FP_ * 2);  // Q [c][h][t][128]
  bf16*  kbuf  = (bf16*) take((long)C_ * HEAD_ * T_ * FP_ * 2);  // K [c][h][t][128]
  bf16*  vtbuf = (bf16*) take((long)C_ * HEAD_ * FP_ * T_ * 2);  // V^T [c][h][128][t]
  float* sc    = (float*)take((long)HEAD_ * T_ * T_ * 4);        // scores (1 country)
  bf16*  at    = (bf16*) take((long)HEAD_ * T_ * T_ * 2);        // probs  (1 country)
  bf16*  ob    = (bf16*) take(CT * HFP_ * 2);                    // concat heads
  float* rs    = (float*)take(CT * F_ * 4);                      // pre-LN residual
  bf16*  hb    = (bf16*) take(CT * FFN_ * 2);                    // FFN hidden
  bf16*  wqt   = (bf16*) take((long)HEAD_ * FP_ * FP_ * 2);
  bf16*  wkt   = (bf16*) take((long)HEAD_ * FP_ * FP_ * 2);
  bf16*  wvt   = (bf16*) take((long)HEAD_ * FP_ * FP_ * 2);
  bf16*  wmt   = (bf16*) take((long)FP_ * HFP_ * 2);
  bf16*  w1t   = (bf16*) take((long)FFN_ * FP_ * 2);
  bf16*  w2t   = (bf16*) take((long)FP_ * FFN_ * 2);

  hipMemcpyAsync(x, X, CT * F_ * sizeof(float), hipMemcpyDeviceToDevice, stream);

  for (int l = 0; l < NLAYERS_; ++l) {
    const long wofs = (long)l * HEAD_ * F_ * F_;
    cast_transpose_pad<<<dim3((FP_ * FP_ + 255) / 256, 1, HEAD_), 256, 0, stream>>>(
        WQ + wofs, wqt, F_, F_, FP_, FP_);
    cast_transpose_pad<<<dim3((FP_ * FP_ + 255) / 256, 1, HEAD_), 256, 0, stream>>>(
        WK + wofs, wkt, F_, F_, FP_, FP_);
    cast_transpose_pad<<<dim3((FP_ * FP_ + 255) / 256, 1, HEAD_), 256, 0, stream>>>(
        WV + wofs, wvt, F_, F_, FP_, FP_);
    cast_wm<<<(FP_ * HFP_ + 255) / 256, 256, 0, stream>>>(Wm + wofs, wmt);
    cast_transpose_pad<<<dim3((FFN_ * FP_ + 255) / 256, 1, 1), 256, 0, stream>>>(
        W1 + (long)l * F_ * FFN_, w1t, F_, FFN_, FP_, FFN_);
    cast_transpose_pad<<<dim3((FP_ * FFN_ + 255) / 256, 1, 1), 256, 0, stream>>>(
        W2 + (long)l * FFN_ * F_, w2t, FFN_, F_, FFN_, FP_);

    cast_pad<<<(unsigned)((CT * FP_ + 255) / 256), 256, 0, stream>>>(
        x, xb, (int)CT, F_, FP_);

    // QKV: z = c*H + h; A batched by country (z/H), B by head (z%H).
    wmma_gemm<EP_BF16><<<dim3(T_ / 128, FP_ / 64, C_ * HEAD_), 256, 0, stream>>>(
        xb, (long)T_ * FP_, HEAD_, FP_, wqt, (long)FP_ * FP_, HEAD_, FP_,
        qbuf, (long)T_ * FP_, FP_, nullptr, nullptr, FP_, FP_, 1.0f);
    wmma_gemm<EP_BF16><<<dim3(T_ / 128, FP_ / 64, C_ * HEAD_), 256, 0, stream>>>(
        xb, (long)T_ * FP_, HEAD_, FP_, wkt, (long)FP_ * FP_, HEAD_, FP_,
        kbuf, (long)T_ * FP_, FP_, nullptr, nullptr, FP_, FP_, 1.0f);
    wmma_gemm<EP_BF16_T><<<dim3(T_ / 128, FP_ / 64, C_ * HEAD_), 256, 0, stream>>>(
        xb, (long)T_ * FP_, HEAD_, FP_, wvt, (long)FP_ * FP_, HEAD_, FP_,
        vtbuf, (long)FP_ * T_, T_, nullptr, nullptr, FP_, FP_, 1.0f);

    // Attention: loop countries (reuse one H*T*T score buffer), batch heads in z.
    for (int c = 0; c < C_; ++c) {
      const bf16* qc = qbuf + (long)c * HEAD_ * T_ * FP_;
      const bf16* kc = kbuf + (long)c * HEAD_ * T_ * FP_;
      const bf16* vc = vtbuf + (long)c * HEAD_ * FP_ * T_;
      // scores = Q K^T * 1/sqrt(100); B-columns of K^T are rows of K (contig).
      wmma_gemm<EP_SCORES><<<dim3(T_ / 128, T_ / 64, HEAD_), 256, 0, stream>>>(
          qc, (long)T_ * FP_, 1, FP_, kc, (long)T_ * FP_, HEAD_, FP_,
          sc, (long)T_ * T_, T_, nullptr, nullptr, FP_, T_, 0.1f);
      softmax_rows<<<HEAD_ * T_ / 8, 256, 0, stream>>>(sc, at);
      // o = attn @ V; B-columns of V are rows of V^T (contig). Store into the
      // per-head slice of the concat buffer (z stride = 128 columns).
      wmma_gemm<EP_BF16><<<dim3(T_ / 128, FP_ / 64, HEAD_), 256, 0, stream>>>(
          at, (long)T_ * T_, 1, T_, vc, (long)FP_ * T_, HEAD_, T_,
          ob + (long)c * T_ * HFP_, (long)FP_, HFP_, nullptr, nullptr, T_, FP_,
          1.0f);
    }

    // x = LN1(x + o @ Wm)
    wmma_gemm<EP_RESID_F32><<<dim3((int)(CT / 128), FP_ / 64, 1), 256, 0, stream>>>(
        ob, 0L, 1, HFP_, wmt, 0L, 1, HFP_, rs, 0L, F_, nullptr, x, HFP_, F_,
        1.0f);
    layernorm_rows<<<(int)(CT / 8), 256, 0, stream>>>(rs, x, LN1g + l * F_,
                                                      LN1b + l * F_);

    // x = LN2(x + relu(x@W1+b1)@W2+b2)
    cast_pad<<<(unsigned)((CT * FP_ + 255) / 256), 256, 0, stream>>>(
        x, xb, (int)CT, F_, FP_);
    wmma_gemm<EP_BIAS_RELU_BF16><<<dim3((int)(CT / 128), FFN_ / 64, 1), 256, 0,
                                   stream>>>(
        xb, 0L, 1, FP_, w1t, 0L, 1, FP_, hb, 0L, FFN_, B1 + (long)l * FFN_,
        nullptr, FP_, FFN_, 1.0f);
    wmma_gemm<EP_BIAS_RESID_F32><<<dim3((int)(CT / 128), FP_ / 64, 1), 256, 0,
                                   stream>>>(
        hb, 0L, 1, FFN_, w2t, 0L, 1, FFN_, rs, 0L, F_, B2 + (long)l * F_, x,
        FFN_, F_, 1.0f);
    layernorm_rows<<<(int)(CT / 8), 256, 0, stream>>>(rs, x, LN2g + l * F_,
                                                      LN2b + l * F_);
  }

  predictor<<<1, 256, 0, stream>>>(x, LPw1, LPb1, LPw2, LPb2, (float*)d_out);
}
